// Block_25323127177341
// MI455X (gfx1250) — matrix operand order repaired and checked by reading, not compiled, not run on hardware
//
#include <hip/hip_runtime.h>
#include <hip/hip_bf16.h>

// ---------------------------------------------------------------------------
// Transformer block (MHA + LN + top-2 MoE + LN + lb_loss) for gfx1250.
// All large matmuls use v_wmma_f32_16x16x32_bf16 (wave32 WMMA, fp32 accum).
// B=4, T=1024, C=1024, H=16 (d=64), E=8, K=2, HID=4096.
// ---------------------------------------------------------------------------

typedef __attribute__((ext_vector_type(16))) __bf16 bf16x16;
typedef __attribute__((ext_vector_type(8)))  float  f32x8;

#define BT_   4096   // B*T
#define C_    1024
#define T_    1024
#define B_    4
#define H_    16
#define D_    64
#define E_    8
#define HID_  4096

union FragU {
    bf16x16 v;
    float4  q[2];
};

__device__ __forceinline__ f32x8 wmma_bf16(bf16x16 a, bf16x16 b, f32x8 c) {
    return __builtin_amdgcn_wmma_f32_16x16x32_bf16(
        /*neg_a=*/false, a, /*neg_b=*/false, b,
        /*c_mod=*/(short)0, c, /*reuse_a=*/false, /*reuse_b=*/false);
}

// A fragment (16x32 bf16). rowp = this lane's row base pointer (row = lane&15).
// Layout: lanes 0-15 hold K=[0..15];
//         lanes 16-31 hold K=[16..31].
__device__ __forceinline__ bf16x16 load_a_frag(const __bf16* rowp, int k_base) {
    int hi = (threadIdx.x >> 4) & 1;
    const __bf16* p = rowp + k_base + (hi << 4);
    FragU f;
    f.q[0] = *(const float4*)(p);
    f.q[1] = *(const float4*)(p + 8);
    return f.v;
}

// B fragment (32x16 bf16) read from Bt = B^T stored row-major [N][K]:
// lane holds column n = n_base + (lane&15), K range = k_base + (lane>>4)*16 + 0..15
// (16 contiguous bf16 = two b128 loads).
__device__ __forceinline__ bf16x16 load_b_frag(const __bf16* Bt, int ldb,
                                               int n_base, int k_base) {
    int l = threadIdx.x & 31;
    const __bf16* p = Bt + (size_t)(n_base + (l & 15)) * ldb + k_base + ((l >> 4) << 4);
    FragU f;
    f.q[0] = ((const float4*)p)[0];
    f.q[1] = ((const float4*)p)[1];
    return f.v;
}

// ---------------------------------------------------------------------------
// Conversion kernels
// ---------------------------------------------------------------------------
__global__ void cvt_bf16(const float* __restrict__ src, __bf16* __restrict__ dst,
                         size_t n) {
    size_t i = (size_t)blockIdx.x * blockDim.x + threadIdx.x;
    size_t stride = (size_t)gridDim.x * blockDim.x;
    for (; i < n; i += stride) dst[i] = (__bf16)src[i];
}

// dst[c*R + r] = src[r*Cc + c], batched over blockIdx.z (matrix stride R*Cc).
__global__ void cvtT_bf16(const float* __restrict__ src, __bf16* __restrict__ dst,
                          int R, int Cc) {
    size_t mat = (size_t)R * Cc;
    const float* s = src + (size_t)blockIdx.z * mat;
    __bf16*      d = dst + (size_t)blockIdx.z * mat;
    size_t i = (size_t)blockIdx.x * blockDim.x + threadIdx.x;
    size_t stride = (size_t)gridDim.x * blockDim.x;
    for (; i < mat; i += stride) {
        size_t r = i / Cc, c = i % Cc;
        d[c * (size_t)R + r] = (__bf16)s[i];
    }
}

__global__ void zero_i32(int* p, int n) {
    int i = blockIdx.x * blockDim.x + threadIdx.x;
    if (i < n) p[i] = 0;
}

// ---------------------------------------------------------------------------
// Generic GEMM: out[M,N] = A[M,K] @ Bt^T + bias.  One wave = 16x64 tile.
// mode 0: fp32 row-major; mode 1: bf16 row-major; mode 2: bf16 [b][n][t] (V^T).
// ---------------------------------------------------------------------------
__global__ __launch_bounds__(32)
void gemm_bf16_16x64(const __bf16* __restrict__ A, const __bf16* __restrict__ Bt,
                     const float* __restrict__ bias, void* __restrict__ outp,
                     int Kdim, int lda, int ldb, int ldo, int mode) {
    int m_base = blockIdx.x << 4;
    int n_base = blockIdx.y << 6;
    int l = threadIdx.x & 31, colL = l & 15, hi = l >> 4;
    const __bf16* rowp = A + (size_t)(m_base + colL) * lda;

    f32x8 c0 = {0,0,0,0,0,0,0,0}, c1 = c0, c2 = c0, c3 = c0;
    for (int k = 0; k < Kdim; k += 32) {
        __builtin_prefetch(rowp + k + 128, 0, 1);  // global_prefetch_b8
        bf16x16 a = load_a_frag(rowp, k);
        c0 = wmma_bf16(a, load_b_frag(Bt, ldb, n_base,      k), c0);
        c1 = wmma_bf16(a, load_b_frag(Bt, ldb, n_base + 16, k), c1);
        c2 = wmma_bf16(a, load_b_frag(Bt, ldb, n_base + 32, k), c2);
        c3 = wmma_bf16(a, load_b_frag(Bt, ldb, n_base + 48, k), c3);
    }
    f32x8 cc[4] = {c0, c1, c2, c3};
    #pragma unroll
    for (int j = 0; j < 4; j++) {
        int n = n_base + j * 16 + colL;
        float bv = bias ? bias[n] : 0.f;
        #pragma unroll
        for (int i = 0; i < 8; i++) {
            int m = m_base + hi * 8 + i;   // C/D layout: row = hi*8 + i
            float v = cc[j][i] + bv;
            if (mode == 0) {
                ((float*)outp)[(size_t)m * ldo + n] = v;
            } else if (mode == 1) {
                ((__bf16*)outp)[(size_t)m * ldo + n] = (__bf16)v;
            } else {  // V transposed: m = b*T + t  ->  out[b][n][t]
                int bb = m >> 10, tt = m & 1023;
                ((__bf16*)outp)[((size_t)bb * C_ + n) * T_ + tt] = (__bf16)v;
            }
        }
    }
}

// ---------------------------------------------------------------------------
// Flash attention: one wave per (16-row q tile, head, batch). d=64.
// S = Q K^T via WMMA; online softmax; P staged in LDS to repack as A-frag;
// O += P V via WMMA (V stored time-major so B-frags are contiguous).
// ---------------------------------------------------------------------------
__global__ __launch_bounds__(32)
void attn_flash(const __bf16* __restrict__ Qb, const __bf16* __restrict__ Kb,
                const __bf16* __restrict__ Vt, const int* __restrict__ amask,
                __bf16* __restrict__ Ob) {
    int qt = blockIdx.x, h = blockIdx.y, b = blockIdx.z;
    int l = threadIdx.x & 31, colL = l & 15, hi = l >> 4;

    const __bf16* Q  = Qb + (size_t)b * T_ * C_ + h * D_;
    const __bf16* Kp = Kb + (size_t)b * T_ * C_ + h * D_;
    const __bf16* Vp = Vt + ((size_t)b * C_ + h * D_) * T_;

    const __bf16* qrow = Q + (size_t)(qt * 16 + colL) * C_;
    bf16x16 aq0 = load_a_frag(qrow, 0);
    bf16x16 aq1 = load_a_frag(qrow, 32);

    f32x8 o0 = {0,0,0,0,0,0,0,0}, o1 = o0, o2 = o0, o3 = o0;
    float mrow[8], lrow[8];
    #pragma unroll
    for (int i = 0; i < 8; i++) { mrow[i] = -1e30f; lrow[i] = 0.f; }

    __shared__ __align__(16) __bf16 Plds[16 * 32];

    int nblk = ((qt * 16 + 15) >> 5) + 1;   // causal: 32-key blocks
    for (int j = 0; j < nblk; j++) {
        int k0 = j << 5;
        f32x8 s0 = {0,0,0,0,0,0,0,0}, s1 = s0;
        s0 = wmma_bf16(aq0, load_b_frag(Kp, C_, k0,      0),  s0);
        s0 = wmma_bf16(aq1, load_b_frag(Kp, C_, k0,      32), s0);
        s1 = wmma_bf16(aq0, load_b_frag(Kp, C_, k0 + 16, 0),  s1);
        s1 = wmma_bf16(aq1, load_b_frag(Kp, C_, k0 + 16, 32), s1);

        int kc0 = k0 + colL, kc1 = k0 + 16 + colL;
        bool am0 = amask[b * T_ + kc0] > 0;
        bool am1 = amask[b * T_ + kc1] > 0;

        #pragma unroll
        for (int i = 0; i < 8; i++) {
            int qr = qt * 16 + hi * 8 + i;
            float v0 = (am0 && kc0 <= qr) ? s0[i] * 0.125f : -1e9f;
            float v1 = (am1 && kc1 <= qr) ? s1[i] * 0.125f : -1e9f;
            // row max across the 16 lanes holding this row's 32 columns
            float rm = fmaxf(fmaxf(v0, v1), mrow[i]);
            rm = fmaxf(rm, __shfl_xor(rm, 1, 16));
            rm = fmaxf(rm, __shfl_xor(rm, 2, 16));
            rm = fmaxf(rm, __shfl_xor(rm, 4, 16));
            rm = fmaxf(rm, __shfl_xor(rm, 8, 16));
            float alpha = __expf(mrow[i] - rm);
            float p0 = __expf(v0 - rm), p1 = __expf(v1 - rm);
            Plds[(hi * 8 + i) * 32 + colL]      = (__bf16)p0;
            Plds[(hi * 8 + i) * 32 + 16 + colL] = (__bf16)p1;
            float ps = p0 + p1;
            ps += __shfl_xor(ps, 1, 16);
            ps += __shfl_xor(ps, 2, 16);
            ps += __shfl_xor(ps, 4, 16);
            ps += __shfl_xor(ps, 8, 16);
            lrow[i] = lrow[i] * alpha + ps;
            mrow[i] = rm;
            o0[i] *= alpha; o1[i] *= alpha; o2[i] *= alpha; o3[i] *= alpha;
        }
        __syncthreads();
        bf16x16 ap = load_a_frag(Plds + (size_t)colL * 32, 0);  // P 16x32
        o0 = wmma_bf16(ap, load_b_frag(Vp, T_, 0,  k0), o0);
        o1 = wmma_bf16(ap, load_b_frag(Vp, T_, 16, k0), o1);
        o2 = wmma_bf16(ap, load_b_frag(Vp, T_, 32, k0), o2);
        o3 = wmma_bf16(ap, load_b_frag(Vp, T_, 48, k0), o3);
        __syncthreads();
    }

    f32x8 oo[4] = {o0, o1, o2, o3};
    #pragma unroll
    for (int j = 0; j < 4; j++)
        #pragma unroll
        for (int i = 0; i < 8; i++) {
            int t = qt * 16 + hi * 8 + i;
            float v = oo[j][i] / lrow[i];
            Ob[((size_t)b * T_ + t) * C_ + h * D_ + j * 16 + colL] = (__bf16)v;
        }
}

// ---------------------------------------------------------------------------
// out = LayerNorm(a + b [+ c]) * g + beta  (one 256-thread block per row)
// ---------------------------------------------------------------------------
__global__ __launch_bounds__(256)
void add_ln_kernel(const float* __restrict__ a, const float* __restrict__ b,
                   const float* __restrict__ c, const float* __restrict__ g,
                   const float* __restrict__ be, float* __restrict__ outf,
                   __bf16* __restrict__ outb) {
    int row = blockIdx.x, tid = threadIdx.x;
    size_t base = (size_t)row * C_;
    __shared__ float red[256];
    float v[4];
    float s = 0.f;
    #pragma unroll
    for (int j = 0; j < 4; j++) {
        int ci = tid + j * 256;
        float x = a[base + ci] + b[base + ci];
        if (c) x += c[base + ci];
        v[j] = x; s += x;
    }
    red[tid] = s; __syncthreads();
    for (int o = 128; o > 0; o >>= 1) { if (tid < o) red[tid] += red[tid + o]; __syncthreads(); }
    float mean = red[0] * (1.f / C_);
    __syncthreads();
    s = 0.f;
    #pragma unroll
    for (int j = 0; j < 4; j++) { float d = v[j] - mean; s += d * d; }
    red[tid] = s; __syncthreads();
    for (int o = 128; o > 0; o >>= 1) { if (tid < o) red[tid] += red[tid + o]; __syncthreads(); }
    float rstd = rsqrtf(red[0] * (1.f / C_) + 1e-5f);
    #pragma unroll
    for (int j = 0; j < 4; j++) {
        int ci = tid + j * 256;
        float y = (v[j] - mean) * rstd * g[ci] + be[ci];
        outf[base + ci] = y;
        if (outb) outb[base + ci] = (__bf16)y;
    }
}

// ---------------------------------------------------------------------------
// Router: logits[t,e] = x1[t,:] @ Wr[:,e] + br[e]   (tiny; one thread per (t,e))
// ---------------------------------------------------------------------------
__global__ void router_kernel(const float* __restrict__ x1, const float* __restrict__ Wr,
                              const float* __restrict__ br, float* __restrict__ logits) {
    int idx = blockIdx.x * blockDim.x + threadIdx.x;
    if (idx >= BT_ * E_) return;
    int t = idx >> 3, e = idx & 7;
    float acc = br[e];
    for (int c = 0; c < C_; c++) acc += x1[(size_t)t * C_ + c] * Wr[c * E_ + e];
    logits[idx] = acc;
}

// Top-2 routing: gates (softmax over the 2 selected logits), per-expert lists
// of packed entries (t*2 + slot). Counts are exact (int atomics).
__global__ void route_topk(const float* __restrict__ logits, float* __restrict__ gate,
                           int* __restrict__ list, int* __restrict__ cnt) {
    int t = blockIdx.x * blockDim.x + threadIdx.x;
    if (t >= BT_) return;
    float l[E_];
    #pragma unroll
    for (int e = 0; e < E_; e++) l[e] = logits[t * E_ + e];
    int e0 = 0; float b0 = l[0];
    #pragma unroll
    for (int e = 1; e < E_; e++) if (l[e] > b0) { b0 = l[e]; e0 = e; }
    int e1 = (e0 == 0) ? 1 : 0; float b1 = l[e1];
    #pragma unroll
    for (int e = 0; e < E_; e++)
        if (e != e0 && l[e] > b1) { b1 = l[e]; e1 = e; }
    float ex = __expf(b1 - b0);          // b0 >= b1
    float inv = 1.f / (1.f + ex);
    gate[t * 2 + 0] = inv;
    gate[t * 2 + 1] = ex * inv;
    int p0 = atomicAdd(&cnt[e0], 1); list[e0 * BT_ + p0] = t * 2 + 0;
    int p1 = atomicAdd(&cnt[e1], 1); list[e1 * BT_ + p1] = t * 2 + 1;
}

// ---------------------------------------------------------------------------
// MoE up-proj: h[slot][tok] = relu(x1b[tok] @ W1[e] + b1[e])   (gathered GEMM)
// ---------------------------------------------------------------------------
__global__ __launch_bounds__(32)
void moe_mlp1(const __bf16* __restrict__ X1b, const __bf16* __restrict__ W1t,
              const float* __restrict__ b1, const int* __restrict__ list,
              const int* __restrict__ cnt, __bf16* __restrict__ H) {
    int e = blockIdx.z;
    int m_base = blockIdx.x << 4;
    int n_base = blockIdx.y << 6;
    int ce = cnt[e];
    if (m_base >= ce) return;
    int l = threadIdx.x & 31, colL = l & 15, hi = l >> 4;
    const int* lp = list + e * BT_;
    int ra = m_base + colL; if (ra >= ce) ra = ce - 1;
    const __bf16* rowp = X1b + (size_t)(lp[ra] >> 1) * C_;
    const __bf16* Bt = W1t + (size_t)e * HID_ * C_;

    f32x8 c0 = {0,0,0,0,0,0,0,0}, c1 = c0, c2 = c0, c3 = c0;
    for (int k = 0; k < C_; k += 32) {
        bf16x16 a = load_a_frag(rowp, k);
        c0 = wmma_bf16(a, load_b_frag(Bt, C_, n_base,      k), c0);
        c1 = wmma_bf16(a, load_b_frag(Bt, C_, n_base + 16, k), c1);
        c2 = wmma_bf16(a, load_b_frag(Bt, C_, n_base + 32, k), c2);
        c3 = wmma_bf16(a, load_b_frag(Bt, C_, n_base + 48, k), c3);
    }
    f32x8 cc[4] = {c0, c1, c2, c3};
    #pragma unroll
    for (int i = 0; i < 8; i++) {
        int m = m_base + hi * 8 + i;
        if (m >= ce) continue;
        int ent = lp[m];
        __bf16* hrow = H + ((size_t)(ent & 1) * BT_ + (ent >> 1)) * HID_;
        #pragma unroll
        for (int j = 0; j < 4; j++) {
            int n = n_base + j * 16 + colL;
            float v = cc[j][i] + b1[e * HID_ + n];
            hrow[n] = (__bf16)fmaxf(v, 0.f);
        }
    }
}

// MoE down-proj: part[slot][tok] = gate * (h[slot][tok] @ W2[e] + b2[e])
__global__ __launch_bounds__(32)
void moe_mlp2(const __bf16* __restrict__ H, const __bf16* __restrict__ W2t,
              const float* __restrict__ b2, const float* __restrict__ gate,
              const int* __restrict__ list, const int* __restrict__ cnt,
              float* __restrict__ part) {
    int e = blockIdx.z;
    int m_base = blockIdx.x << 4;
    int n_base = blockIdx.y << 6;
    int ce = cnt[e];
    if (m_base >= ce) return;
    int l = threadIdx.x & 31, colL = l & 15, hi = l >> 4;
    const int* lp = list + e * BT_;
    int ra = m_base + colL; if (ra >= ce) ra = ce - 1;
    int entA = lp[ra];
    const __bf16* rowp = H + ((size_t)(entA & 1) * BT_ + (entA >> 1)) * HID_;
    const __bf16* Bt = W2t + (size_t)e * C_ * HID_;

    f32x8 c0 = {0,0,0,0,0,0,0,0}, c1 = c0, c2 = c0, c3 = c0;
    for (int k = 0; k < HID_; k += 32) {
        bf16x16 a = load_a_frag(rowp, k);
        c0 = wmma_bf16(a, load_b_frag(Bt, HID_, n_base,      k), c0);
        c1 = wmma_bf16(a, load_b_frag(Bt, HID_, n_base + 16, k), c1);
        c2 = wmma_bf16(a, load_b_frag(Bt, HID_, n_base + 32, k), c2);
        c3 = wmma_bf16(a, load_b_frag(Bt, HID_, n_base + 48, k), c3);
    }
    f32x8 cc[4] = {c0, c1, c2, c3};
    #pragma unroll
    for (int i = 0; i < 8; i++) {
        int m = m_base + hi * 8 + i;
        if (m >= ce) continue;
        int ent = lp[m];
        int tok = ent >> 1, slot = ent & 1;
        float gt = gate[tok * 2 + slot];
        float* prow = part + ((size_t)slot * BT_ + tok) * C_;
        #pragma unroll
        for (int j = 0; j < 4; j++) {
            int n = n_base + j * 16 + colL;
            prow[n] = gt * (cc[j][i] + b2[e * C_ + n]);
        }
    }
}

// ---------------------------------------------------------------------------
// lb_loss = E * sum_e importance_e * load_e   (single block, fixed-order tree)
// ---------------------------------------------------------------------------
__global__ __launch_bounds__(256)
void lb_loss_kernel(const float* __restrict__ logits, const int* __restrict__ cnt,
                    float* __restrict__ out) {
    __shared__ float red[256];
    int tid = threadIdx.x;
    float s[E_] = {0, 0, 0, 0, 0, 0, 0, 0};
    for (int t = tid; t < BT_; t += 256) {
        float l[E_]; float m = -1e30f;
        #pragma unroll
        for (int e = 0; e < E_; e++) { l[e] = logits[t * E_ + e]; m = fmaxf(m, l[e]); }
        float z = 0.f;
        #pragma unroll
        for (int e = 0; e < E_; e++) { l[e] = __expf(l[e] - m); z += l[e]; }
        float iz = 1.f / z;
        #pragma unroll
        for (int e = 0; e < E_; e++) s[e] += l[e] * iz;
    }
    float lb = 0.f;
    for (int e = 0; e < E_; e++) {
        red[tid] = s[e]; __syncthreads();
        for (int o = 128; o > 0; o >>= 1) { if (tid < o) red[tid] += red[tid + o]; __syncthreads(); }
        if (tid == 0) {
            float imp  = red[0] / (float)BT_;
            float load = (float)cnt[e] / (float)BT_;
            lb += imp * load;
        }
        __syncthreads();
    }
    if (tid == 0) out[0] = (float)E_ * lb;
}

// ---------------------------------------------------------------------------
// Host launcher
// ---------------------------------------------------------------------------
extern "C" void kernel_launch(void* const* d_in, const int* in_sizes, int n_in,
                              void* d_out, int out_size, void* d_ws, size_t ws_size,
                              hipStream_t stream) {
    (void)in_sizes; (void)n_in; (void)out_size; (void)ws_size;
    const float* x    = (const float*)d_in[0];
    const int*   am   = (const int*)  d_in[1];
    const float* Wq   = (const float*)d_in[2];  const float* bq  = (const float*)d_in[3];
    const float* Wk   = (const float*)d_in[4];  const float* bk  = (const float*)d_in[5];
    const float* Wv   = (const float*)d_in[6];  const float* bv  = (const float*)d_in[7];
    const float* Wo   = (const float*)d_in[8];  const float* bo  = (const float*)d_in[9];
    const float* ln1g = (const float*)d_in[10]; const float* ln1b = (const float*)d_in[11];
    const float* ln2g = (const float*)d_in[12]; const float* ln2b = (const float*)d_in[13];
    const float* Wr   = (const float*)d_in[14]; const float* br  = (const float*)d_in[15];
    const float* W1   = (const float*)d_in[16]; const float* b1  = (const float*)d_in[17];
    const float* W2   = (const float*)d_in[18]; const float* b2  = (const float*)d_in[19];

    char* w = (char*)d_ws;
    size_t off = 0;
    auto take = [&](size_t bytes) -> char* {
        char* r = w + off;
        off += (bytes + 255) & ~(size_t)255;
        return r;
    };
    __bf16* XB   = (__bf16*)take((size_t)BT_ * C_ * 2);
    __bf16* WQT  = (__bf16*)take((size_t)C_ * C_ * 2);
    __bf16* WKT  = (__bf16*)take((size_t)C_ * C_ * 2);
    __bf16* WVT  = (__bf16*)take((size_t)C_ * C_ * 2);
    __bf16* WOT  = (__bf16*)take((size_t)C_ * C_ * 2);
    __bf16* QB   = (__bf16*)take((size_t)BT_ * C_ * 2);
    __bf16* KB   = (__bf16*)take((size_t)BT_ * C_ * 2);
    __bf16* VTb  = (__bf16*)take((size_t)BT_ * C_ * 2);
    __bf16* OB   = (__bf16*)take((size_t)BT_ * C_ * 2);
    float*  PROJ = (float*) take((size_t)BT_ * C_ * 4);
    float*  X1   = (float*) take((size_t)BT_ * C_ * 4);
    __bf16* X1B  = (__bf16*)take((size_t)BT_ * C_ * 2);
    float*  LOG  = (float*) take((size_t)BT_ * E_ * 4);
    float*  GATE = (float*) take((size_t)BT_ * 2 * 4);
    int*    LIST = (int*)   take((size_t)E_ * BT_ * 4);
    int*    CNT  = (int*)   take((size_t)E_ * 4);
    __bf16* W1T  = (__bf16*)take((size_t)E_ * HID_ * C_ * 2);
    __bf16* W2T  = (__bf16*)take((size_t)E_ * C_ * HID_ * 2);
    __bf16* Hh   = (__bf16*)take((size_t)2 * BT_ * HID_ * 2);
    float*  PART = (float*) take((size_t)2 * BT_ * C_ * 4);

    // 0) zero routing counts (must be re-zeroed every call: deterministic)
    zero_i32<<<1, 32, 0, stream>>>(CNT, E_);

    // 1) bf16 conversions (weights transposed so B-fragments are contiguous)
    cvt_bf16<<<2048, 256, 0, stream>>>(x, XB, (size_t)BT_ * C_);
    cvtT_bf16<<<dim3(1024, 1, 1), 256, 0, stream>>>(Wq, WQT, C_, C_);
    cvtT_bf16<<<dim3(1024, 1, 1), 256, 0, stream>>>(Wk, WKT, C_, C_);
    cvtT_bf16<<<dim3(1024, 1, 1), 256, 0, stream>>>(Wv, WVT, C_, C_);
    cvtT_bf16<<<dim3(1024, 1, 1), 256, 0, stream>>>(Wo, WOT, C_, C_);
    cvtT_bf16<<<dim3(2048, 1, E_), 256, 0, stream>>>(W1, W1T, C_, HID_);   // [C,HID]->[HID,C]
    cvtT_bf16<<<dim3(2048, 1, E_), 256, 0, stream>>>(W2, W2T, HID_, C_);   // [HID,C]->[C,HID]

    // 2) QKV projections (WMMA); V written time-major for attention B-frags
    gemm_bf16_16x64<<<dim3(256, 16), 32, 0, stream>>>(XB, WQT, bq, QB,  C_, C_, C_, C_, 1);
    gemm_bf16_16x64<<<dim3(256, 16), 32, 0, stream>>>(XB, WKT, bk, KB,  C_, C_, C_, C_, 1);
    gemm_bf16_16x64<<<dim3(256, 16), 32, 0, stream>>>(XB, WVT, bv, VTb, C_, C_, C_, C_, 2);

    // 3) flash attention (WMMA for QK^T and PV)
    attn_flash<<<dim3(T_ / 16, H_, B_), 32, 0, stream>>>(QB, KB, VTb, am, OB);

    // 4) output projection + residual + LN1 (also emit bf16 x1 for MoE)
    gemm_bf16_16x64<<<dim3(256, 16), 32, 0, stream>>>(OB, WOT, bo, PROJ, C_, C_, C_, C_, 0);
    add_ln_kernel<<<BT_, 256, 0, stream>>>(x, PROJ, nullptr, ln1g, ln1b, X1, X1B);

    // 5) router + top-2 routing
    router_kernel<<<(BT_ * E_) / 256, 256, 0, stream>>>(X1, Wr, br, LOG);
    route_topk<<<BT_ / 256, 256, 0, stream>>>(LOG, GATE, LIST, CNT);

    // 6) MoE expert MLPs — only routed (token,slot) pairs (2/8 of dense work)
    moe_mlp1<<<dim3(256, HID_ / 64, E_), 32, 0, stream>>>(X1B, W1T, b1, LIST, CNT, Hh);
    moe_mlp2<<<dim3(256, C_ / 64, E_),   32, 0, stream>>>(Hh, W2T, b2, GATE, LIST, CNT, PART);

    // 7) combine + residual + LN2 -> d_out, then lb_loss scalar
    add_ln_kernel<<<BT_, 256, 0, stream>>>(X1, PART, PART + (size_t)BT_ * C_,
                                           ln2g, ln2b, (float*)d_out, nullptr);
    lb_loss_kernel<<<1, 256, 0, stream>>>(LOG, CNT, (float*)d_out + (size_t)BT_ * C_);
}
